// GNT_35605278883936
// MI455X (gfx1250) — matrix operand order repaired and checked by reading, not compiled, not run on hardware
//
#include <hip/hip_runtime.h>
#include <stdint.h>

// ---------------- problem constants (from reference setup) ----------------
#define BATCH 2
#define CIN   64
#define HH    128
#define WW    128
#define HID   16
#define OFC   18
#define NPTS  9
#define PIX   (HH * WW)          // 16384
#define MTOT  (BATCH * PIX)      // 32768
#define KDIM  (NPTS * CIN)       // 576
#define BN_EPS 1e-5f

typedef __attribute__((ext_vector_type(16))) __bf16 v16bf;
typedef __attribute__((ext_vector_type(8)))  float  v8f;

__device__ __forceinline__ uint16_t f2bf(float f) {
  union { float f; uint32_t u; } c; c.f = f;
  uint32_t u = c.u;
  u += 0x7fffu + ((u >> 16) & 1u);   // round-to-nearest-even
  return (uint16_t)(u >> 16);
}

// ---------------- kernel 1: 1x1 conv (64->16) + BN + ReLU ----------------
__global__ __launch_bounds__(256) void k_conv1(
    const float* __restrict__ x, const float* __restrict__ w1,
    const float* __restrict__ g1, const float* __restrict__ b1,
    const float* __restrict__ m1, const float* __restrict__ v1,
    float* __restrict__ f) {
  __shared__ float ws[HID * CIN];
  __shared__ float sc[HID], sb[HID];
  int tid = threadIdx.x;
  for (int i = tid; i < HID * CIN; i += 256) ws[i] = w1[i];
  if (tid < HID) {
    float inv = g1[tid] * rsqrtf(v1[tid] + BN_EPS);
    sc[tid] = inv;
    sb[tid] = b1[tid] - m1[tid] * inv;
  }
  __syncthreads();
  int m = blockIdx.x * 256 + tid;
  int b = m >> 14, p = m & (PIX - 1);
  float acc[HID];
#pragma unroll
  for (int o = 0; o < HID; ++o) acc[o] = 0.f;
  const float* xb = x + (size_t)(b * CIN) * PIX + p;
  for (int ic = 0; ic < CIN; ++ic) {
    float xv = xb[(size_t)ic * PIX];
#pragma unroll
    for (int o = 0; o < HID; ++o) acc[o] = fmaf(xv, ws[o * CIN + ic], acc[o]);
  }
  float* fb = f + (size_t)(b * HID) * PIX + p;
#pragma unroll
  for (int o = 0; o < HID; ++o) {
    float r = fmaf(acc[o], sc[o], sb[o]);
    fb[(size_t)o * PIX] = r > 0.f ? r : 0.f;
  }
}

// ------- kernel 2: 3x3 conv (16->18)+BN+ReLU fused with 1x1 (18->18)+BN+ReLU -------
__global__ __launch_bounds__(256) void k_offset(
    const float* __restrict__ f, const float* __restrict__ wdw,
    const float* __restrict__ g2, const float* __restrict__ b2,
    const float* __restrict__ m2, const float* __restrict__ v2,
    const float* __restrict__ w2, const float* __restrict__ g3,
    const float* __restrict__ b3, const float* __restrict__ m3,
    const float* __restrict__ v3, float* __restrict__ off) {
  __shared__ float wdws[OFC * HID * NPTS];   // 2592 floats
  __shared__ float w2s[OFC * OFC];           // 324 floats
  __shared__ float s2[OFC], bb2[OFC], s3[OFC], bb3[OFC];
  int tid = threadIdx.x;
  for (int i = tid; i < OFC * HID * NPTS; i += 256) wdws[i] = wdw[i];
  for (int i = tid; i < OFC * OFC; i += 256) w2s[i] = w2[i];
  if (tid < OFC) {
    float inv2 = g2[tid] * rsqrtf(v2[tid] + BN_EPS);
    s2[tid] = inv2; bb2[tid] = b2[tid] - m2[tid] * inv2;
    float inv3 = g3[tid] * rsqrtf(v3[tid] + BN_EPS);
    s3[tid] = inv3; bb3[tid] = b3[tid] - m3[tid] * inv3;
  }
  __syncthreads();
  int m = blockIdx.x * 256 + tid;
  int b = m >> 14, p = m & (PIX - 1);
  int h = p >> 7, w = p & 127;
  float acc[OFC];
#pragma unroll
  for (int o = 0; o < OFC; ++o) acc[o] = 0.f;
  const float* fb = f + (size_t)(b * HID) * PIX;
  for (int ic = 0; ic < HID; ++ic) {
    float fv[NPTS];
#pragma unroll
    for (int n = 0; n < NPTS; ++n) {
      int dx = n / 3 - 1, dy = n % 3 - 1;
      int hh = h + dx, ww = w + dy;
      bool ok = (hh >= 0) & (hh < HH) & (ww >= 0) & (ww < WW);
      fv[n] = ok ? fb[(size_t)ic * PIX + hh * WW + ww] : 0.f;
    }
#pragma unroll
    for (int n = 0; n < NPTS; ++n) {
#pragma unroll
      for (int o = 0; o < OFC; ++o)
        acc[o] = fmaf(fv[n], wdws[(o * HID + ic) * NPTS + n], acc[o]);
    }
  }
  float t[OFC];
#pragma unroll
  for (int o = 0; o < OFC; ++o) {
    float r = fmaf(acc[o], s2[o], bb2[o]);
    t[o] = r > 0.f ? r : 0.f;
  }
  float* ob = off + (size_t)m * OFC;
#pragma unroll
  for (int o = 0; o < OFC; ++o) {
    float s = 0.f;
#pragma unroll
    for (int j = 0; j < OFC; ++j) s = fmaf(w2s[o * OFC + j], t[j], s);
    float r = fmaf(s, s3[o], bb3[o]);
    ob[o] = r > 0.f ? r : 0.f;
  }
}

// ------- kernel 3: repack w_out into bf16 WMMA B-fragment images -------
// WBfrag[((i*4+t)*32+l)*16 + j] = WB[k][oc] with
//   k = i*32 + (l>=16 ? 16 : 0) + j   (K order: k = n*64 + ic, matches A tile)
//   oc = t*16 + (l & 15)
// WB[n*64+ic][oc] = w_out[oc][ic][n/3][n%3] = w_out_flat[(oc*64+ic)*9 + n]
__global__ __launch_bounds__(256) void k_wprep(
    const float* __restrict__ wout, uint16_t* __restrict__ wb) {
  int idx = blockIdx.x * 256 + threadIdx.x;     // 18*4*32*16 = 36864 total
  if (idx >= 18 * 4 * 32 * 16) return;
  int j = idx & 15;
  int l = (idx >> 4) & 31;
  int t = (idx >> 9) & 3;
  int i = idx >> 11;
  int k = i * 32 + ((l >= 16) ? 16 : 0) + j;
  int oc = t * 16 + (l & 15);
  int n = k >> 6;
  int ic = k & 63;
  wb[idx] = f2bf(wout[(oc * 64 + ic) * 9 + n]);
}

// ------- kernel 4: bilinear sampling into LDS A-tiles + WMMA GEMM + residual -------
#define WAVES 4
#define TPB (WAVES * 32)
#define LDS_WB_BYTES  (18 * 4 * 32 * 16 * 2)          // 73728
#define LDS_A_BYTES   (WAVES * 16 * KDIM * 2)         // 73728
#define LDS_TOTAL     (LDS_WB_BYTES + LDS_A_BYTES)    // 147456

union FragU { uint4 q[2]; v16bf v; };

__global__ __launch_bounds__(TPB) void k_deform_gemm(
    const float* __restrict__ x, const float* __restrict__ off,
    const uint16_t* __restrict__ wbf, float* __restrict__ out) {
  extern __shared__ char lds[];
  uint32_t* wb32 = (uint32_t*)lds;                    // 18432 dwords
  uint32_t* a32  = (uint32_t*)(lds + LDS_WB_BYTES);   // 18432 dwords
  const int tid  = threadIdx.x;
  const int lane = tid & 31;
  const int wid  = tid >> 5;

  // phase 1: stage repacked B fragments into LDS
  const uint32_t* wbsrc = (const uint32_t*)wbf;
  for (int i = tid; i < LDS_WB_BYTES / 4; i += TPB) wb32[i] = wbsrc[i];

  // phase 2: bilinear sampling -> per-wave A tile (16 rows x 576 K, bf16)
  const int m0 = blockIdx.x * 64 + wid * 16;   // 64-aligned, never crosses batch
  const int b  = m0 >> 14;
  const float* xb = x + (size_t)(b * CIN) * PIX;
  uint32_t* aw = a32 + wid * (16 * KDIM / 2);  // 4608 dwords per wave
  const int ic0 = lane * 2;
  const float* ch0 = xb + (size_t)ic0 * PIX;
  const float* ch1 = ch0 + PIX;

  for (int r = 0; r < 16; ++r) {
    int m = m0 + r;
    int p = m & (PIX - 1);
    int h = p >> 7, w = p & 127;
    const float* om = off + (size_t)m * OFC;
#pragma unroll
    for (int n = 0; n < NPTS; ++n) {
      float ox = om[n], oy = om[NPTS + n];
      float px = (float)(h + 1) + (float)(n / 3 - 1) + ox;
      float py = (float)(w + 1) + (float)(n % 3 - 1) + oy;
      float flx = floorf(px), fly = floorf(py);
      float qltx = fminf(fmaxf(flx, 0.f), 129.f);
      float qlty = fminf(fmaxf(fly, 0.f), 129.f);
      float qrbx = fminf(fmaxf(flx + 1.f, 0.f), 129.f);
      float qrby = fminf(fmaxf(fly + 1.f, 0.f), 129.f);
      px = fminf(fmaxf(px, 0.f), 129.f);
      py = fminf(fmaxf(py, 0.f), 129.f);
      float glt = (1.f + (qltx - px)) * (1.f + (qlty - py));
      float grb = (1.f - (qrbx - px)) * (1.f - (qrby - py));
      float glb = (1.f + (qltx - px)) * (1.f - (qrby - py));
      float grt = (1.f - (qrbx - px)) * (1.f + (qlty - py));
      int ilx = (int)qltx, ily = (int)qlty, irx = (int)qrbx, iry = (int)qrby;
      // padded(130x130, zero border) -> unpadded x
      bool bxl = (ilx >= 1) & (ilx <= 128);
      bool bxr = (irx >= 1) & (irx <= 128);
      bool byl = (ily >= 1) & (ily <= 128);
      bool byr = (iry >= 1) & (iry <= 128);
      int o_lt = ((ilx - 1) << 7) + (ily - 1);
      int o_rb = ((irx - 1) << 7) + (iry - 1);
      int o_lb = ((ilx - 1) << 7) + (iry - 1);
      int o_rt = ((irx - 1) << 7) + (ily - 1);
      float v0 = 0.f, v1 = 0.f;
      if (bxl & byl) { float a = ch0[o_lt], c = ch1[o_lt]; v0 = fmaf(glt, a, v0); v1 = fmaf(glt, c, v1); }
      if (bxr & byr) { float a = ch0[o_rb], c = ch1[o_rb]; v0 = fmaf(grb, a, v0); v1 = fmaf(grb, c, v1); }
      if (bxl & byr) { float a = ch0[o_lb], c = ch1[o_lb]; v0 = fmaf(glb, a, v0); v1 = fmaf(glb, c, v1); }
      if (bxr & byl) { float a = ch0[o_rt], c = ch1[o_rt]; v0 = fmaf(grt, a, v0); v1 = fmaf(grt, c, v1); }
      uint32_t packed = (uint32_t)f2bf(v0) | ((uint32_t)f2bf(v1) << 16);
      aw[r * (KDIM / 2) + n * 32 + lane] = packed;   // k = n*64 + ic0
    }
  }
  __syncthreads();

  // phase 3: K-loop GEMM, D(16x64 per wave) = A(16x576) x WB(576x64)
  v8f cacc[4] = {};
  const int arow = lane & 15;
  const int akb  = (lane < 16) ? 0 : 16;             // contiguous A K-base (halfs)
  const uint32_t abase = wid * 4608 + arow * (KDIM / 2) + (akb >> 1);
  for (int i = 0; i < KDIM / 32; ++i) {              // 18 K-steps of 32
    FragU a;
    const uint32_t* ap = a32 + abase + i * 16;
    a.q[0] = *(const uint4*)(ap);                    // K = ks+kb .. +7
    a.q[1] = *(const uint4*)(ap + 4);                // K = ks+kb+8 .. +15
#pragma unroll
    for (int t = 0; t < 4; ++t) {
      FragU bfr;
      const uint32_t* bp = wb32 + ((i * 4 + t) * 32 + lane) * 8;
      bfr.q[0] = *(const uint4*)(bp);
      bfr.q[1] = *(const uint4*)(bp + 4);
      cacc[t] = __builtin_amdgcn_wmma_f32_16x16x32_bf16(
          false, a.v, false, bfr.v, (short)0, cacc[t], false, false);
    }
  }

  // store with fused residual: out[b,oc,h,w] = x[b,oc,h,w] + D
  const int rbase = (lane >= 16) ? 8 : 0;
#pragma unroll
  for (int t = 0; t < 4; ++t) {
    int oc = t * 16 + (lane & 15);
#pragma unroll
    for (int r8 = 0; r8 < 8; ++r8) {
      int m = m0 + rbase + r8;
      int p = m & (PIX - 1);
      size_t idx = (size_t)(b * CIN + oc) * PIX + p;
      out[idx] = x[idx] + cacc[t][r8];
    }
  }
}

// ---------------- launcher ----------------
extern "C" void kernel_launch(void* const* d_in, const int* in_sizes, int n_in,
                              void* d_out, int out_size, void* d_ws, size_t ws_size,
                              hipStream_t stream) {
  const float* x   = (const float*)d_in[0];
  const float* w1  = (const float*)d_in[1];
  const float* g1  = (const float*)d_in[2];
  const float* b1  = (const float*)d_in[3];
  const float* m1  = (const float*)d_in[4];
  const float* v1  = (const float*)d_in[5];
  const float* wdw = (const float*)d_in[6];
  const float* g2  = (const float*)d_in[7];
  const float* b2  = (const float*)d_in[8];
  const float* m2  = (const float*)d_in[9];
  const float* v2  = (const float*)d_in[10];
  const float* w2  = (const float*)d_in[11];
  const float* g3  = (const float*)d_in[12];
  const float* b3  = (const float*)d_in[13];
  const float* m3  = (const float*)d_in[14];
  const float* v3  = (const float*)d_in[15];
  const float* wout= (const float*)d_in[16];
  float* out = (float*)d_out;

  char* ws = (char*)d_ws;
  float*    f_buf   = (float*)ws;                                  // 2,097,152 B
  float*    off_buf = (float*)(ws + 2097152);                      // 2,359,296 B
  uint16_t* wb_buf  = (uint16_t*)(ws + 2097152 + 2359296);         //    73,728 B

  (void)hipFuncSetAttribute(reinterpret_cast<const void*>(k_deform_gemm),
                            hipFuncAttributeMaxDynamicSharedMemorySize, LDS_TOTAL);

  k_conv1<<<MTOT / 256, 256, 0, stream>>>(x, w1, g1, b1, m1, v1, f_buf);
  k_offset<<<MTOT / 256, 256, 0, stream>>>(f_buf, wdw, g2, b2, m2, v2,
                                           w2, g3, b3, m3, v3, off_buf);
  k_wprep<<<(18 * 4 * 32 * 16) / 256, 256, 0, stream>>>(wout, wb_buf);
  k_deform_gemm<<<MTOT / 64, TPB, LDS_TOTAL, stream>>>(x, off_buf, wb_buf, out);
}
